// MixtureOfExperts_56495999811834
// MI455X (gfx1250) — matrix operand order repaired and checked by reading, not compiled, not run on hardware
//
#include <hip/hip_runtime.h>
#include <math.h>

// ---------------- problem constants ----------------
#define B_DIM   4
#define S_DIM   2048
#define N_TOK   (B_DIM * S_DIM)   // 8192 tokens
#define D_DIM   1024
#define H_DIM   4096
#define E_NUM   8

#define M_TILE  16
#define XSTRIDE (D_DIM + 8)       // bf16 elems; 1032*2 = 2064 B row stride (16B aligned)
#define HSTRIDE (H_DIM + 8)       // 4104*2 = 8208 B (16B aligned)
#define TPB     512               // 16 waves per block

typedef __attribute__((ext_vector_type(16))) __bf16 v16bf;
typedef __attribute__((ext_vector_type(8)))  float  v8f;
typedef __attribute__((ext_vector_type(4)))  unsigned int u32x4;
typedef __attribute__((ext_vector_type(8)))  int         i32x8;
typedef __attribute__((ext_vector_type(4)))  int         i32x4;

// float -> bf16 bits, round-to-nearest-even
__device__ __forceinline__ unsigned short f2bf(float f) {
    union { float f; unsigned int u; } v; v.f = f;
    unsigned int r = v.u + 0x7FFFu + ((v.u >> 16) & 1u);
    return (unsigned short)(r >> 16);
}

// branchless tanh-approx GELU: 0.5*v*(1+tanh(u)) == v*sigmoid(2u)
__device__ __forceinline__ float gelu_fast(float v) {
    float z = 1.5957691216057308f * (v + 0.044715f * v * v * v);   // 2u
    float e = __builtin_amdgcn_exp2f(-1.4426950408889634f * z);
    return v * __builtin_amdgcn_rcpf(1.0f + e);
}

__device__ __forceinline__ v8f wmma_bf16(uint4 a0, uint4 a1, uint4 b0, uint4 b1, v8f c) {
    union { v16bf v; uint4 u[2]; } A, B;
    A.u[0] = a0; A.u[1] = a1; B.u[0] = b0; B.u[1] = b1;
    return __builtin_amdgcn_wmma_f32_16x16x32_bf16(false, A.v, false, B.v,
                                                   (short)0, c, false, false);
}

// ---------------- kernel 0: zero expert counters ----------------
__global__ void zero_counts_kernel(int* __restrict__ counts) {
    if (threadIdx.x < E_NUM) counts[threadIdx.x] = 0;
}

// ---------------- kernel 1: fp32 -> bf16 transposed weight conversion ----
__global__ void transpose_convert_kernel(const float* __restrict__ in,
                                         unsigned short* __restrict__ out,
                                         int R, int C) {
    __shared__ float tile[32][33];
    int tilesC = C >> 5;
    int tpb    = (R >> 5) * tilesC;
    int b   = blockIdx.x / tpb;
    int rem = blockIdx.x % tpb;
    int r0  = (rem / tilesC) << 5;
    int c0  = (rem % tilesC) << 5;
    const float* bin = in + (size_t)b * R * C;
    unsigned short* bout = out + (size_t)b * R * C;
    for (int i = threadIdx.y; i < 32; i += 8)
        tile[i][threadIdx.x] = bin[(size_t)(r0 + i) * C + (c0 + threadIdx.x)];
    __syncthreads();
    for (int i = threadIdx.y; i < 32; i += 8)
        bout[(size_t)(c0 + i) * R + (r0 + threadIdx.x)] = f2bf(tile[threadIdx.x][i]);
}

// ---------------- kernel 2: router (one wave per token) ----------------
__global__ void router_kernel(const float* __restrict__ x,
                              const float* __restrict__ rW,   // [D,E]
                              const float* __restrict__ rb,   // [E]
                              float* __restrict__ gate,       // [N,E]
                              int*   __restrict__ counts,     // [E]
                              int*   __restrict__ lists)      // [E,N] tok<<1|slot
{
    int wave  = threadIdx.x >> 5;
    int lane  = threadIdx.x & 31;
    int token = blockIdx.x * 8 + wave;
    if (token >= N_TOK) return;

    float p[E_NUM];
#pragma unroll
    for (int e = 0; e < E_NUM; ++e) p[e] = 0.0f;

    const float* xr = x + (size_t)token * D_DIM;
    for (int d = lane; d < D_DIM; d += 32) {
        float xv = xr[d];
        const float4* w = (const float4*)(rW + (size_t)d * E_NUM);
        float4 w0 = w[0], w1 = w[1];
        p[0] += xv * w0.x; p[1] += xv * w0.y; p[2] += xv * w0.z; p[3] += xv * w0.w;
        p[4] += xv * w1.x; p[5] += xv * w1.y; p[6] += xv * w1.z; p[7] += xv * w1.w;
    }
#pragma unroll
    for (int off = 16; off > 0; off >>= 1)
#pragma unroll
        for (int e = 0; e < E_NUM; ++e)
            p[e] += __shfl_down(p[e], off, 32);

    if (lane == 0) {
#pragma unroll
        for (int e = 0; e < E_NUM; ++e) p[e] += rb[e];
        float mx = p[0];
#pragma unroll
        for (int e = 1; e < E_NUM; ++e) mx = fmaxf(mx, p[e]);
        float s = 0.0f;
#pragma unroll
        for (int e = 0; e < E_NUM; ++e) { p[e] = __expf(p[e] - mx); s += p[e]; }
        float inv = 1.0f / s;
#pragma unroll
        for (int e = 0; e < E_NUM; ++e) p[e] *= inv;

        int i0 = 0;
#pragma unroll
        for (int e = 1; e < E_NUM; ++e) if (p[e] > p[i0]) i0 = e;
        int i1 = (i0 == 0) ? 1 : 0;
#pragma unroll
        for (int e = 0; e < E_NUM; ++e) if (e != i0 && p[e] > p[i1]) i1 = e;

        float g0 = p[i0], g1 = p[i1], gs = g0 + g1;
        g0 /= gs; g1 /= gs;
#pragma unroll
        for (int e = 0; e < E_NUM; ++e)
            gate[(size_t)token * E_NUM + e] = (e == i0) ? g0 : ((e == i1) ? g1 : 0.0f);

        int pos0 = atomicAdd(&counts[i0], 1);
        lists[(size_t)i0 * N_TOK + pos0] = (token << 1);
        int pos1 = atomicAdd(&counts[i1], 1);
        lists[(size_t)i1 * N_TOK + pos1] = (token << 1) | 1;
    }
}

// ---------------- kernel 3: fused routed expert FC1+GELU+FC2 -------------
// One 512-thread (16-wave) block per (expert, 16-token tile).
// X tile gathered via TDM gather-mode tensor_load_to_lds (16 row indices).
__global__ void __launch_bounds__(TPB)
moe_expert_kernel(const float* __restrict__ x,
                  const float* __restrict__ b1,           // [E,H]
                  const float* __restrict__ b2,           // [E,D]
                  const unsigned short* __restrict__ W1t, // [E][H][D] bf16
                  const unsigned short* __restrict__ W2t, // [E][D][H] bf16
                  const float* __restrict__ gate,         // [N,E]
                  const int*   __restrict__ counts,
                  const int*   __restrict__ lists,
                  float* __restrict__ out2)               // [2][N][D]
{
    extern __shared__ char smem[];
    int*            tokS = (int*)smem;                                   // 16
    float*          gS   = (float*)(smem + 64);                          // 16
    unsigned short* Xs   = (unsigned short*)(smem + 128);                // 16 x XSTRIDE bf16
    unsigned short* Hs   = (unsigned short*)(smem + 128 + M_TILE * XSTRIDE * 2);
    float*          Xf   = (float*)(smem + 128 + M_TILE * XSTRIDE * 2
                                         + M_TILE * HSTRIDE * 2);        // 16 x 1024 fp32

    const int MAX_TILES = N_TOK / M_TILE;     // 512
    int e  = blockIdx.x / MAX_TILES;
    int t  = blockIdx.x % MAX_TILES;
    int Me = counts[e];
    if (t * M_TILE >= Me) return;

    int tid  = threadIdx.x;
    int wave = tid >> 5;                      // 0..15
    int lane = tid & 31;
    int half = lane >> 4;                     // 0: lanes 0-15, 1: lanes 16-31
    int l16  = lane & 15;

    if (tid < M_TILE) {
        int r  = t * M_TILE + tid;
        int pk = (r < Me) ? lists[(size_t)e * N_TOK + r] : -1;
        tokS[tid] = pk;
        gS[tid]   = (pk >= 0) ? gate[(size_t)(pk >> 1) * E_NUM + e] : 0.0f;
    }
    __syncthreads();

    // ---- TDM: gather 16 token rows (fp32, 1024 elems each) into Xf ----
    if (wave == 0) {
        int rows[M_TILE];
#pragma unroll
        for (int r = 0; r < M_TILE; ++r) {
            int pk = tokS[r];
            rows[r] = (pk >= 0) ? (pk >> 1) : 0;
        }
        unsigned long long ga = (unsigned long long)(uintptr_t)x;
        unsigned int lds_off  = (unsigned int)(uintptr_t)Xf;

        u32x4 g0;
        g0[0] = 1u | (1u << 31);                      // count=1, gather_mode=1, 16-bit idx
        g0[1] = lds_off;                              // lds_addr
        g0[2] = (unsigned int)(ga & 0xFFFFFFFFull);   // global_addr[31:0]
        g0[3] = (unsigned int)((ga >> 32) & 0x01FFFFFFull) | (2u << 30); // addr hi | type=2

        i32x8 g1;
        g1[0] = (int)(2u << 16);                      // data_size = 2 (4 bytes)
        g1[1] = (int)((D_DIM & 0xFFFF) << 16);        // tensor_dim0[15:0] in [31:16]
        g1[2] = (int)(((unsigned)D_DIM >> 16) | ((N_TOK & 0xFFFF) << 16)); // dim0 hi | dim1 lo
        g1[3] = (int)(((unsigned)N_TOK >> 16) | ((D_DIM & 0xFFFF) << 16)); // dim1 hi | tile_dim0
        g1[4] = M_TILE;                               // tile_dim1 = #indices, tile_dim2 = 0
        g1[5] = D_DIM;                                // tensor_dim0_stride[31:0]
        g1[6] = 0;                                    // stride hi | dim1_stride lo (ignored)
        g1[7] = 0;

        i32x4 g2, g3;
#pragma unroll
        for (int i = 0; i < 4; ++i) {
            g2[i] = (rows[2 * i] & 0xFFFF) | ((rows[2 * i + 1] & 0xFFFF) << 16);
            g3[i] = (rows[8 + 2 * i] & 0xFFFF) | ((rows[9 + 2 * i] & 0xFFFF) << 16);
        }
        i32x8 g4 = {};                                // extra group (6-arg form): zero-filled
        __builtin_amdgcn_tensor_load_to_lds(g0, g1, g2, g3, g4, 0);
        __builtin_amdgcn_s_wait_tensorcnt(0);
    }
    __syncthreads();

    // ---- convert Xf fp32 -> Xs bf16 (padded stride) ----
    for (int idx = tid; idx < (M_TILE * D_DIM) / 4; idx += TPB) {
        int r  = idx / (D_DIM / 4);
        int c4 = (idx % (D_DIM / 4)) * 4;
        const float4 v = *(const float4*)(Xf + r * D_DIM + c4);
        unsigned short* dst = Xs + r * XSTRIDE + c4;
        dst[0] = f2bf(v.x); dst[1] = f2bf(v.y); dst[2] = f2bf(v.z); dst[3] = f2bf(v.w);
    }
    __syncthreads();

    // ---- FC1: H[16x4096] = gelu(X @ W1 + b1); 16 waves x 32 cols per sweep ----
    const unsigned short* arowX = Xs + l16 * XSTRIDE + half * 16;
    for (int hc = 0; hc < H_DIM; hc += TPB) {          // 8 sweeps of 512 cols
        int ncol0 = hc + wave * 32 + l16;
        int ncol1 = ncol0 + 16;
        const unsigned short* wc0 = W1t + ((size_t)e * H_DIM + ncol0) * D_DIM + half * 16;
        const unsigned short* wc1 = W1t + ((size_t)e * H_DIM + ncol1) * D_DIM + half * 16;
        v8f acc0 = {}, acc1 = {};
        uint4 a0 = *(const uint4*)(arowX);
        uint4 a1 = *(const uint4*)(arowX + 8);
        uint4 p0 = *(const uint4*)(wc0);
        uint4 p1 = *(const uint4*)(wc0 + 8);
        uint4 q0 = *(const uint4*)(wc1);
        uint4 q1 = *(const uint4*)(wc1 + 8);
        for (int k = 0; k < D_DIM; k += 32) {
            int kn = (k + 32 < D_DIM) ? (k + 32) : 0;  // branchless wrap prefetch
            __builtin_prefetch(wc0 + kn + 256, 0, 1);
            uint4 na0 = *(const uint4*)(arowX + kn);
            uint4 na1 = *(const uint4*)(arowX + kn + 8);
            uint4 np0 = *(const uint4*)(wc0 + kn);
            uint4 np1 = *(const uint4*)(wc0 + kn + 8);
            uint4 nq0 = *(const uint4*)(wc1 + kn);
            uint4 nq1 = *(const uint4*)(wc1 + kn + 8);
            acc0 = wmma_bf16(a0, a1, p0, p1, acc0);
            acc1 = wmma_bf16(a0, a1, q0, q1, acc1);
            a0 = na0; a1 = na1; p0 = np0; p1 = np1; q0 = nq0; q1 = nq1;
        }
        float bias0 = b1[(size_t)e * H_DIM + ncol0];
        float bias1 = b1[(size_t)e * H_DIM + ncol1];
#pragma unroll
        for (int r = 0; r < 8; ++r) {
            int row = r + half * 8;
            Hs[row * HSTRIDE + ncol0] = f2bf(gelu_fast(acc0[r] + bias0));
            Hs[row * HSTRIDE + ncol1] = f2bf(gelu_fast(acc1[r] + bias1));
        }
    }
    __syncthreads();

    // ---- FC2: OUT[16x1024] = (H @ W2 + b2) * gate; 16 waves x 64 cols each ----
    const unsigned short* arowH = Hs + l16 * HSTRIDE + half * 16;
    for (int s = 0; s < 2; ++s) {                      // 2 dual-slice sweeps
        int ncol0 = wave * 64 + s * 32 + l16;
        int ncol1 = ncol0 + 16;
        const unsigned short* wc0 = W2t + ((size_t)e * D_DIM + ncol0) * H_DIM + half * 16;
        const unsigned short* wc1 = W2t + ((size_t)e * D_DIM + ncol1) * H_DIM + half * 16;
        v8f acc0 = {}, acc1 = {};
        uint4 a0 = *(const uint4*)(arowH);
        uint4 a1 = *(const uint4*)(arowH + 8);
        uint4 p0 = *(const uint4*)(wc0);
        uint4 p1 = *(const uint4*)(wc0 + 8);
        uint4 q0 = *(const uint4*)(wc1);
        uint4 q1 = *(const uint4*)(wc1 + 8);
        for (int k = 0; k < H_DIM; k += 32) {
            int kn = (k + 32 < H_DIM) ? (k + 32) : 0;
            __builtin_prefetch(wc0 + kn + 256, 0, 1);
            uint4 na0 = *(const uint4*)(arowH + kn);
            uint4 na1 = *(const uint4*)(arowH + kn + 8);
            uint4 np0 = *(const uint4*)(wc0 + kn);
            uint4 np1 = *(const uint4*)(wc0 + kn + 8);
            uint4 nq0 = *(const uint4*)(wc1 + kn);
            uint4 nq1 = *(const uint4*)(wc1 + kn + 8);
            acc0 = wmma_bf16(a0, a1, p0, p1, acc0);
            acc1 = wmma_bf16(a0, a1, q0, q1, acc1);
            a0 = na0; a1 = na1; p0 = np0; p1 = np1; q0 = nq0; q1 = nq1;
        }
        float bias0 = b2[(size_t)e * D_DIM + ncol0];
        float bias1 = b2[(size_t)e * D_DIM + ncol1];
#pragma unroll
        for (int r = 0; r < 8; ++r) {
            int row = r + half * 8;
            int pk  = tokS[row];
            if (pk >= 0) {
                int tok  = pk >> 1;
                int slot = pk & 1;
                float* o = out2 + ((size_t)slot * N_TOK + tok) * D_DIM;
                o[ncol0] = (acc0[r] + bias0) * gS[row];
                o[ncol1] = (acc1[r] + bias1) * gS[row];
            }
        }
    }
}

// ---------------- kernel 4: combine slot0 + slot1, write aux loss --------
__global__ void combine_kernel(const float* __restrict__ out2,
                               float* __restrict__ out) {
    size_t i  = (size_t)blockIdx.x * blockDim.x + threadIdx.x;
    size_t n4 = (size_t)N_TOK * D_DIM / 4;
    if (i < n4) {
        const float4* a = (const float4*)out2;
        const float4* b = (const float4*)(out2 + (size_t)N_TOK * D_DIM);
        float4 va = a[i], vb = b[i];
        float4 r; r.x = va.x + vb.x; r.y = va.y + vb.y;
        r.z = va.z + vb.z; r.w = va.w + vb.w;
        ((float4*)out)[i] = r;
    }
    if (i == 0) out[(size_t)N_TOK * D_DIM] = 0.0f;   // aux_loss
}

// ---------------- launch ----------------
extern "C" void kernel_launch(void* const* d_in, const int* in_sizes, int n_in,
                              void* d_out, int out_size, void* d_ws, size_t ws_size,
                              hipStream_t stream) {
    const float* x  = (const float*)d_in[0];
    const float* rW = (const float*)d_in[1];
    const float* rb = (const float*)d_in[2];
    const float* W1 = (const float*)d_in[3];
    const float* b1 = (const float*)d_in[4];
    const float* W2 = (const float*)d_in[5];
    const float* b2 = (const float*)d_in[6];
    float* out = (float*)d_out;

    char* ws = (char*)d_ws;
    size_t szW = (size_t)E_NUM * D_DIM * H_DIM * sizeof(unsigned short); // 67 MB each
    unsigned short* W1t = (unsigned short*)ws;
    unsigned short* W2t = (unsigned short*)(ws + szW);
    float* gate = (float*)(ws + 2 * szW);
    int*   counts = (int*)(ws + 2 * szW + (size_t)N_TOK * E_NUM * sizeof(float));
    int*   lists  = (int*)((char*)counts + 256);
    float* out2   = (float*)((char*)lists + (size_t)E_NUM * N_TOK * sizeof(int));

    zero_counts_kernel<<<1, 32, 0, stream>>>(counts);
    transpose_convert_kernel<<<E_NUM * (D_DIM / 32) * (H_DIM / 32), dim3(32, 8), 0, stream>>>(
        W1, W1t, D_DIM, H_DIM);
    transpose_convert_kernel<<<E_NUM * (H_DIM / 32) * (D_DIM / 32), dim3(32, 8), 0, stream>>>(
        W2, W2t, H_DIM, D_DIM);
    router_kernel<<<N_TOK / 8, 256, 0, stream>>>(x, rW, rb, gate, counts, lists);

    int smem = 128 + M_TILE * XSTRIDE * 2 + M_TILE * HSTRIDE * 2
                   + M_TILE * D_DIM * (int)sizeof(float);            // 230,016 B
    (void)hipFuncSetAttribute(reinterpret_cast<const void*>(moe_expert_kernel),
                              hipFuncAttributeMaxDynamicSharedMemorySize, smem);
    moe_expert_kernel<<<E_NUM * (N_TOK / M_TILE), TPB, smem, stream>>>(
        x, b1, b2, W1t, W2t, gate, counts, lists, out2);

    combine_kernel<<<((N_TOK * D_DIM / 4) + 255) / 256, 256, 0, stream>>>(out2, out);
}
